// VectorQuantizer_30631706755894
// MI455X (gfx1250) — compile-verified, matrix operand order checked by reading, not run on hardware
//
#include <hip/hip_runtime.h>
#include <hip/hip_bf16.h>
#include <math.h>

typedef __attribute__((ext_vector_type(16))) __bf16 v16bf;
typedef __attribute__((ext_vector_type(8)))  float  v8f;

#define N_ROWS 8192   // B*S = 8*1024
#define KCODES 8192
#define DIM    256
#define SLEN   1024
#define QELEMS (N_ROWS * DIM)

#define NCH    64     // codes per block chunk
#define NLDS   32     // codes of each chunk staged in LDS (rest read via L2)
#define LSTR   264    // padded LDS row stride (elements): bank-quad advance of 4

// dynamic LDS bytes: 2 planes x 2 buffers x NLDS*LSTR bf16 + wnorm[KCODES] f32
#define SMEM_BYTES (4 * NLDS * LSTR * 2 + KCODES * 4)

union BF16Frag {
    v16bf v;
    uint4 q[2];
    unsigned short u[16];
};

// ---------------------------------------------------------------------------
// K0: row norms for x and W; split W into bf16 hi/lo planes (truncate-split).
// ---------------------------------------------------------------------------
__global__ __launch_bounds__(256) void vq_prep(
    const float* __restrict__ x, const float* __restrict__ W,
    unsigned short* __restrict__ w_hi, unsigned short* __restrict__ w_lo,
    float* __restrict__ xnorm, float* __restrict__ wnorm)
{
    __shared__ float red[256];
    const int row = blockIdx.x;
    const int t = threadIdx.x;
    const bool isW = row >= N_ROWS;
    const float* src = isW ? (W + (size_t)(row - N_ROWS) * DIM)
                           : (x + (size_t)row * DIM);
    float val = src[t];
    if (isW) {
        unsigned int bits = __float_as_uint(val);
        float hif = __uint_as_float(bits & 0xFFFF0000u);
        float r = val - hif;                                  // exact residual
        size_t o = (size_t)(row - N_ROWS) * DIM + t;
        w_hi[o] = (unsigned short)(bits >> 16);
        w_lo[o] = (unsigned short)(__float_as_uint(r) >> 16);
    }
    red[t] = val * val;
    __syncthreads();
    for (int s = 128; s > 0; s >>= 1) {
        if (t < s) red[t] += red[t + s];
        __syncthreads();
    }
    if (t == 0) {
        if (isW) wnorm[row - N_ROWS] = red[0];
        else     xnorm[row] = red[0];
    }
}

// ---------------------------------------------------------------------------
// K1: distance GEMM (bf16x3 WMMA, f32 acc) + per-row running argmin.
// 64 blocks x 256 threads (8 waves, wave owns 16 rows). Per 64-code chunk:
//   - codes [0,32): async-copied global->LDS (double buffered, ASYNCcnt)
//   - codes [32,64): read directly from global (L2-resident W planes)
// 4 independent accumulators (one per 16-code sub-tile) for WMMA ILP.
// LDS plane layout (all offsets computed arithmetically -- no LDS-derived
// pointer aggregates, those become illegal static initializers):
//   [ h(buf0) | h(buf1) | l(buf0) | l(buf1) | wnorm ]
// ---------------------------------------------------------------------------
__device__ __forceinline__ unsigned short* lds_plane(unsigned short* smem,
                                                     int plane, int bufi)
{
    return smem + (plane * 2 + bufi) * (NLDS * LSTR);
}

__global__ __launch_bounds__(256) void vq_gemm_argmin(
    const float* __restrict__ x,
    const unsigned short* __restrict__ w_hi,
    const unsigned short* __restrict__ w_lo,
    const float* __restrict__ xnorm, const float* __restrict__ wnorm,
    float* __restrict__ dist_out, int* __restrict__ idx_out)
{
    extern __shared__ unsigned short smem_u16[];
    float* s_wn = (float*)(smem_u16 + 4 * NLDS * LSTR);

    for (int i = threadIdx.x; i < KCODES; i += 256) s_wn[i] = wnorm[i];

    const int wave    = threadIdx.x >> 5;
    const int lane    = threadIdx.x & 31;
    const int laneHi  = lane >> 4;
    const int lane16  = lane & 15;
    const int rowBase = blockIdx.x * 128 + wave * 16;
    const int rOff    = laneHi ? 8 : 0;

    // Async-fill of one chunk's LDS stage: NLDS codes x 256 d x 2 planes,
    // 8 x b128 per thread, via the CDNA5 async-to-LDS path (ASYNCcnt).
    auto issue_async = [&](int cb, int bufi) {
        const int nBase = cb * NCH;
#pragma unroll
        for (int i = 0; i < 8; ++i) {
            int e     = i * 256 + threadIdx.x;   // 0..2047
            int plane = e >> 10;
            int r     = e & 1023;
            int c     = r >> 5;                  // local code 0..31
            int s8    = r & 31;                  // 8-elem segment 0..31
            const unsigned short* g =
                (plane ? w_lo : w_hi) + (size_t)(nBase + c) * DIM + s8 * 8;
            unsigned short* l = lds_plane(smem_u16, plane, bufi) + (c * LSTR + s8 * 8);
            unsigned dsaddr = (unsigned)(unsigned long long)(void*)l;
            asm volatile("global_load_async_to_lds_b128 %0, %1, off"
                         :: "v"(dsaddr), "v"(g) : "memory");
        }
    };

    // --- A fragments (bf16 hi/lo) built once; resident in VGPRs.
    BF16Frag ahi[8], alo[8];
#pragma unroll
    for (int s = 0; s < 8; ++s) {
        const float* src = x + (size_t)(rowBase + lane16) * DIM + s * 32 + rOff;
        float4 f0 = *(const float4*)(src);
        float4 f1 = *(const float4*)(src + 4);
        float4 f2 = *(const float4*)(src + 16);
        float4 f3 = *(const float4*)(src + 20);
        float buf[16] = { f0.x, f0.y, f0.z, f0.w, f1.x, f1.y, f1.z, f1.w,
                          f2.x, f2.y, f2.z, f2.w, f3.x, f3.y, f3.z, f3.w };
#pragma unroll
        for (int h = 0; h < 16; ++h) {
            unsigned int bits = __float_as_uint(buf[h]);
            ahi[s].u[h] = (unsigned short)(bits >> 16);
            float hif = __uint_as_float(bits & 0xFFFF0000u);
            float r = buf[h] - hif;
            alo[s].u[h] = (unsigned short)(__float_as_uint(r) >> 16);
        }
    }

    float best[8];
    int   bidx[8];
#pragma unroll
    for (int v = 0; v < 8; ++v) { best[v] = 3.4e38f; bidx[v] = 0; }

    // Prologue: stage chunk 0, wait ASYNCcnt, then block barrier.
    issue_async(0, 0);
    asm volatile("s_wait_asynccnt 0x0" ::: "memory");
    __syncthreads();

    int buf = 0;
    for (int cb = 0; cb < KCODES / NCH; ++cb) {
        const int nBase = cb * NCH;
        if (cb + 1 < KCODES / NCH) issue_async(cb + 1, buf ^ 1);

        const unsigned short* hb = lds_plane(smem_u16, 0, buf);
        const unsigned short* lb = lds_plane(smem_u16, 1, buf);

        v8f c[4] = {};
#pragma unroll
        for (int s = 0; s < 8; ++s) {
#pragma unroll
            for (int jn = 0; jn < 4; ++jn) {
                const int cl = jn * 16 + lane16;
                BF16Frag bh, bl;
                if (jn < 2) {           // LDS-staged codes (DS pipe)
                    const unsigned short* ph = hb + cl * LSTR + s * 32 + rOff;
                    const unsigned short* pl = lb + cl * LSTR + s * 32 + rOff;
                    bh.q[0] = *(const uint4*)(ph);
                    bh.q[1] = *(const uint4*)(ph + 16);
                    bl.q[0] = *(const uint4*)(pl);
                    bl.q[1] = *(const uint4*)(pl + 16);
                } else {                // L2-resident codes (VMEM pipe)
                    const unsigned short* ph =
                        w_hi + (size_t)(nBase + cl) * DIM + s * 32 + rOff;
                    const unsigned short* pl =
                        w_lo + (size_t)(nBase + cl) * DIM + s * 32 + rOff;
                    bh.q[0] = *(const uint4*)(ph);
                    bh.q[1] = *(const uint4*)(ph + 16);
                    bl.q[0] = *(const uint4*)(pl);
                    bl.q[1] = *(const uint4*)(pl + 16);
                }
                // bf16x3: hi*hi + hi*lo + lo*hi, f32 accumulate
                c[jn] = __builtin_amdgcn_wmma_f32_16x16x32_bf16(
                            false, ahi[s].v, false, bh.v, (short)0, c[jn], false, false);
                c[jn] = __builtin_amdgcn_wmma_f32_16x16x32_bf16(
                            false, ahi[s].v, false, bl.v, (short)0, c[jn], false, false);
                c[jn] = __builtin_amdgcn_wmma_f32_16x16x32_bf16(
                            false, alo[s].v, false, bh.v, (short)0, c[jn], false, false);
            }
        }

#pragma unroll
        for (int jn = 0; jn < 4; ++jn) {
            const int n = nBase + jn * 16 + lane16;
            const float wn = s_wn[n];
#pragma unroll
            for (int v = 0; v < 8; ++v) {
                float val = wn - 2.0f * c[jn][v];   // |x|^2 added at the end
                if (val < best[v]) { best[v] = val; bidx[v] = n; }
            }
        }

        asm volatile("s_wait_asynccnt 0x0" ::: "memory");
        __syncthreads();
        buf ^= 1;
    }

    // Reduce (min,argmin) across the 16 lanes of each half-wave group.
#pragma unroll
    for (int v = 0; v < 8; ++v) {
        float bv = best[v];
        int   bi = bidx[v];
#pragma unroll
        for (int mask = 8; mask >= 1; mask >>= 1) {
            float ov = __shfl_xor(bv, mask, 32);
            int   oi = __shfl_xor(bi, mask, 32);
            if (ov < bv || (ov == bv && oi < bi)) { bv = ov; bi = oi; }
        }
        if (lane16 == 0) {
            int row = rowBase + v + 8 * laneHi;
            dist_out[row] = bv + xnorm[row];
            idx_out[row]  = bi;
        }
    }
}

// ---------------------------------------------------------------------------
// K2: stable per-batch rank sort (== jnp.argsort) + faithful reindexing bug:
// new[b*S + rank(i)] = enc_flat[i] with i in [0,S) (batch-0 region!).
// ---------------------------------------------------------------------------
__global__ __launch_bounds__(1024) void vq_sortperm(
    const float* __restrict__ dist, const int* __restrict__ idx_in,
    int* __restrict__ idx_out)
{
    __shared__ float sv[SLEN];
    const int b = blockIdx.x, i = threadIdx.x;
    const float v = dist[b * SLEN + i];
    sv[i] = v;
    __syncthreads();
    int rank = 0;
    for (int j = 0; j < SLEN; ++j) {
        float u = sv[j];
        rank += (u < v) || (u == v && j < i);
    }
    idx_out[b * SLEN + rank] = idx_in[i];   // gather from flat[0..S) — reference bug
}

// ---------------------------------------------------------------------------
// K3: quantized = W[idx]; accumulate sum((q - x)^2). One row per block.
// ---------------------------------------------------------------------------
__global__ __launch_bounds__(256) void vq_gather_loss(
    const float* __restrict__ x, const float* __restrict__ W,
    const int* __restrict__ idx, float* __restrict__ out_quant,
    float* __restrict__ accum)
{
    __shared__ float red[256];
    const int n = blockIdx.x, t = threadIdx.x;
    const int k = idx[n];
    float q  = W[(size_t)k * DIM + t];
    float xi = x[(size_t)n * DIM + t];
    out_quant[(size_t)n * DIM + t] = q;     // quantized_st == quantized (fwd)
    float d = q - xi;
    red[t] = d * d;
    __syncthreads();
    for (int s = 128; s > 0; s >>= 1) {
        if (t < s) red[t] += red[t + s];
        __syncthreads();
    }
    if (t == 0) atomicAdd(accum, red[0]);
}

// ---------------------------------------------------------------------------
// Small kernels: zero scratch, histogram (+index output), entropy, finalize.
// ---------------------------------------------------------------------------
__global__ void vq_zero(float* counts, float* accum)
{
    int i = blockIdx.x * 256 + threadIdx.x;
    if (i < KCODES) counts[i] = 0.0f;
    if (i < 16)     accum[i]  = 0.0f;
}

__global__ void vq_hist(const int* __restrict__ idx, float* __restrict__ counts,
                        float* __restrict__ out_idx)
{
    int n = blockIdx.x * 256 + threadIdx.x;
    if (n < N_ROWS) {
        int k = idx[n];
        atomicAdd(&counts[k], 1.0f);
        out_idx[n] = (float)k;
    }
}

__global__ void vq_entropy(const float* __restrict__ counts, float* __restrict__ accum)
{
    int k = blockIdx.x * 256 + threadIdx.x;
    if (k < KCODES) {
        float p = counts[k] * (1.0f / (float)N_ROWS);
        atomicAdd(&accum[1], -p * logf(p + 1e-10f));
    }
}

__global__ void vq_final(const float* __restrict__ accum, float* __restrict__ out)
{
    if (threadIdx.x == 0 && blockIdx.x == 0) {
        // loss = q_latent + 0.99*e_latent = 1.99 * mse(q, x)
        out[0] = accum[0] * (1.99f / (float)QELEMS);
        out[1 + QELEMS] = expf(accum[1]);   // perplexity
    }
}

// ---------------------------------------------------------------------------
extern "C" void kernel_launch(void* const* d_in, const int* in_sizes, int n_in,
                              void* d_out, int out_size, void* d_ws, size_t ws_size,
                              hipStream_t stream)
{
    const float* x = (const float*)d_in[0];   // [8,1024,256] f32
    const float* W = (const float*)d_in[1];   // [8192,256]   f32
    float* out = (float*)d_out;               // [1 | 2097152 | 1 | 8192] as f32

    char* ws = (char*)d_ws;
    unsigned short* w_hi = (unsigned short*)ws;                       // 4 MB
    unsigned short* w_lo = (unsigned short*)(ws + (4ull << 20));      // 4 MB
    float* xnorm = (float*)(ws + (8ull << 20));                       // 32 KB
    float* wnorm = xnorm + KCODES;
    float* dist  = wnorm + KCODES;
    int*   enc   = (int*)(dist + N_ROWS);
    int*   encp  = enc + N_ROWS;
    float* counts = (float*)(encp + N_ROWS);
    float* accum  = counts + KCODES;

    vq_zero       <<<32, 256, 0, stream>>>(counts, accum);
    vq_prep       <<<N_ROWS + KCODES, 256, 0, stream>>>(x, W, w_hi, w_lo, xnorm, wnorm);
    vq_gemm_argmin<<<64, 256, SMEM_BYTES, stream>>>(x, w_hi, w_lo, xnorm, wnorm, dist, enc);
    vq_sortperm   <<<8, 1024, 0, stream>>>(dist, enc, encp);
    vq_gather_loss<<<N_ROWS, 256, 0, stream>>>(x, W, encp, out + 1, accum);
    vq_hist       <<<32, 256, 0, stream>>>(encp, counts, out + 2 + QELEMS);
    vq_entropy    <<<32, 256, 0, stream>>>(counts, accum);
    vq_final      <<<1, 1, 0, stream>>>(accum, out);
}
